// VoxelFeatureExtractor_21543555957029
// MI455X (gfx1250) — matrix operand order repaired and checked by reading, not compiled, not run on hardware
//
#include <hip/hip_runtime.h>

#define DEVINL __device__ __forceinline__

typedef __attribute__((ext_vector_type(16))) _Float16 v16h;
typedef __attribute__((ext_vector_type(2)))  _Float16 h2;
typedef __attribute__((ext_vector_type(8)))  float    v8f;

// ---------------- problem constants ----------------
constexpr int NV  = 50000;
constexpr int T   = 16;
constexpr int CIN = 8;
constexpr int P   = 8;
constexpr int NB  = 120000;
constexpr int OH  = 512;
constexpr int OW  = 512;
constexpr int G   = 16;
constexpr int ROWS = NV * T;            // 800000
constexpr int NTB  = NB / 16;           // 7500 bev tiles (exact)
constexpr float EPS = 1e-5f;

// ---------------- helpers ----------------
DEVINL float mish_f(float x) {
    float sp = (x > 20.f) ? x : log1pf(__expf(x));
    return x * tanhf(sp);
}

DEVINL v8f wmma32(v16h a, v16h b, v8f c) {
    return __builtin_amdgcn_wmma_f32_16x16x32_f16(false, a, false, b, (short)0, c, false, false);
}

// A fragment: 16x32 f16 tile, row-major in LDS with leading dim ldk.
DEVINL v16h frag_A(const _Float16* __restrict__ t, int ldk, int kbase, int lane) {
    int m = lane & 15, hi = lane >> 4;
    v16h r;
#pragma unroll
    for (int v = 0; v < 8; ++v) {
        int k = kbase + ((v < 4) ? 0 : 16) + hi * 8 + (v & 3) * 2;
        r[2 * v]     = t[m * ldk + k];
        r[2 * v + 1] = t[m * ldk + k + 1];
    }
    return r;
}

// B fragment: weights stored [O][C] f16 (leading dim ldc). B[k][n] = W[nbase+n][kbase+k].
DEVINL v16h frag_B(const _Float16* __restrict__ w, int ldc, int nbase, int kbase, int lane) {
    int n = lane & 15, hi = lane >> 4;
    v16h r;
#pragma unroll
    for (int v = 0; v < 8; ++v) {
        int k = kbase + ((v < 4) ? 0 : 16) + hi * 8 + (v & 3) * 2;
        r[2 * v]     = w[(nbase + n) * ldc + k];
        r[2 * v + 1] = w[(nbase + n) * ldc + k + 1];
    }
    return r;
}

// ---- CDNA5 async global<->LDS DMA (ASYNCcnt, ISA §10.7 / §15.18) ----
// Generic->LDS aperture: flat addr low 32 bits == LDS byte offset.
DEVINL unsigned lds_offset(const void* p) {
    return (unsigned)(unsigned long long)p;
}
DEVINL void async_load_b128(unsigned lds_off, const void* gaddr) {
    asm volatile("global_load_async_to_lds_b128 %0, %1, off"
                 :: "v"(lds_off), "v"(gaddr) : "memory");
}
DEVINL void async_store_b128(const void* gaddr, unsigned lds_off) {
    asm volatile("global_store_async_from_lds_b128 %0, %1, off"
                 :: "v"(gaddr), "v"(lds_off) : "memory");
}
DEVINL void wait_async0() { asm volatile("s_wait_asynccnt 0x0" ::: "memory"); }
DEVINL void wait_ds0()    { asm volatile("s_wait_dscnt 0x0"    ::: "memory"); }

// ---------------- zero output ----------------
__global__ __launch_bounds__(256) void k_zero(float* __restrict__ p, int n4) {
    float4 z = {0.f, 0.f, 0.f, 0.f};
    for (int i = blockIdx.x * blockDim.x + threadIdx.x; i < n4; i += gridDim.x * blockDim.x)
        ((float4*)p)[i] = z;
}

// ---------------- generic BN finalize: partials -> (scale,shift) ----------------
__global__ void k_finalize(const float* __restrict__ part, int nwg, int nch, float cnt,
                           const float* __restrict__ gamma, const float* __restrict__ beta,
                           float* __restrict__ st) {
    int c = threadIdx.x;
    if (c >= nch) return;
    float s = 0.f, q = 0.f;
    for (int w = 0; w < nwg; ++w) {
        s += part[w * 2 * nch + c];
        q += part[w * 2 * nch + nch + c];
    }
    float m  = s / cnt;
    float v  = q / cnt - m * m;
    float sc = gamma[c] * rsqrtf(v + EPS);
    st[2 * c]     = sc;
    st[2 * c + 1] = beta[c] - m * sc;
}

// ---------------- S1: stats of y1 = voxels @ W1^T (8 ch) ----------------
__global__ __launch_bounds__(256) void k_s1(const float* __restrict__ vox,
                                            const float* __restrict__ W1,
                                            float* __restrict__ part) {
    __shared__ float sW[64];
    __shared__ float red[8 * 16];
    int tid = threadIdx.x;
    if (tid < 64) sW[tid] = W1[tid];
    __syncthreads();
    float s[8], q[8];
#pragma unroll
    for (int o = 0; o < 8; ++o) { s[o] = 0.f; q[o] = 0.f; }
    for (int r = blockIdx.x * blockDim.x + tid; r < ROWS; r += gridDim.x * blockDim.x) {
        float x[8];
#pragma unroll
        for (int c = 0; c < 8; ++c) x[c] = vox[r * 8 + c];
#pragma unroll
        for (int o = 0; o < 8; ++o) {
            float y = 0.f;
#pragma unroll
            for (int c = 0; c < 8; ++c) y = fmaf(sW[o * 8 + c], x[c], y);
            s[o] += y; q[o] += y * y;
        }
    }
    int lane = tid & 31, wv = tid >> 5;
#pragma unroll
    for (int o = 0; o < 8; ++o)
#pragma unroll
        for (int off = 16; off > 0; off >>= 1) {
            s[o] += __shfl_xor(s[o], off, 32);
            q[o] += __shfl_xor(q[o], off, 32);
        }
    if (lane == 0) {
#pragma unroll
        for (int o = 0; o < 8; ++o) { red[wv * 16 + o] = s[o]; red[wv * 16 + 8 + o] = q[o]; }
    }
    __syncthreads();
    if (tid < 16) {
        float a = 0.f;
        for (int w = 0; w < 8; ++w) a += red[w * 16 + tid];
        part[blockIdx.x * 16 + tid] = a;
    }
}

// ---------------- S2: apply vfe1 (BN+mish+max+concat) -> x1; stats of y2 (32 ch) ----------------
__global__ __launch_bounds__(256) void k_s2(const float* __restrict__ vox,
                                            const float* __restrict__ W1,
                                            const float* __restrict__ st1,
                                            const float* __restrict__ W2,
                                            float* __restrict__ x1,
                                            float* __restrict__ part) {
    __shared__ float sW1[64], sS1[16], sW2[512];
    __shared__ float red[8 * 64];
    int tid = threadIdx.x;
    if (tid < 64) sW1[tid] = W1[tid];
    if (tid < 16) sS1[tid] = st1[tid];
    for (int i = tid; i < 512; i += 256) sW2[i] = W2[i];
    __syncthreads();
    float s[32], q[32];
#pragma unroll
    for (int o = 0; o < 32; ++o) { s[o] = 0.f; q[o] = 0.f; }
    for (int v = blockIdx.x * blockDim.x + tid; v < NV; v += gridDim.x * blockDim.x) {
        const float* vp = vox + v * T * CIN;
        float mx[8];
#pragma unroll
        for (int o = 0; o < 8; ++o) mx[o] = -3.0e38f;
        for (int t = 0; t < T; ++t) {
            float x[8];
#pragma unroll
            for (int c = 0; c < 8; ++c) x[c] = vp[t * 8 + c];
#pragma unroll
            for (int o = 0; o < 8; ++o) {
                float y = 0.f;
#pragma unroll
                for (int c = 0; c < 8; ++c) y = fmaf(sW1[o * 8 + c], x[c], y);
                float pv = mish_f(fmaf(y, sS1[2 * o], sS1[2 * o + 1]));
                mx[o] = fmaxf(mx[o], pv);
            }
        }
        for (int t = 0; t < T; ++t) {
            float x[8], xr[16];
#pragma unroll
            for (int c = 0; c < 8; ++c) x[c] = vp[t * 8 + c];
#pragma unroll
            for (int o = 0; o < 8; ++o) {
                float y = 0.f;
#pragma unroll
                for (int c = 0; c < 8; ++c) y = fmaf(sW1[o * 8 + c], x[c], y);
                xr[o]     = mish_f(fmaf(y, sS1[2 * o], sS1[2 * o + 1]));
                xr[8 + o] = mx[o];
            }
            int r = v * T + t;
#pragma unroll
            for (int i = 0; i < 16; ++i) x1[r * 16 + i] = xr[i];
#pragma unroll
            for (int o = 0; o < 32; ++o) {
                float y = 0.f;
#pragma unroll
                for (int i = 0; i < 16; ++i) y = fmaf(sW2[o * 16 + i], xr[i], y);
                s[o] += y; q[o] += y * y;
            }
        }
    }
    int lane = tid & 31, wv = tid >> 5;
#pragma unroll
    for (int o = 0; o < 32; ++o)
#pragma unroll
        for (int off = 16; off > 0; off >>= 1) {
            s[o] += __shfl_xor(s[o], off, 32);
            q[o] += __shfl_xor(q[o], off, 32);
        }
    if (lane == 0) {
#pragma unroll
        for (int o = 0; o < 32; ++o) { red[wv * 64 + o] = s[o]; red[wv * 64 + 32 + o] = q[o]; }
    }
    __syncthreads();
    if (tid < 64) {
        float a = 0.f;
        for (int w = 0; w < 8; ++w) a += red[w * 64 + tid];
        part[blockIdx.x * 64 + tid] = a;
    }
}

// ---------------- S3: y2 (WMMA) -> BN+mish+max+concat*mask -> x2; y3 (WMMA) stats (64 ch) ----------------
__global__ __launch_bounds__(256) void k_s3(const float* __restrict__ x1,
                                            const float* __restrict__ vmask,
                                            const float* __restrict__ st2,
                                            const float* __restrict__ W2,
                                            const float* __restrict__ W3,
                                            float* __restrict__ x2g,
                                            float* __restrict__ part) {
    __shared__ _Float16 sW2[32 * 32];     // K padded 16->32
    __shared__ _Float16 sW3[64 * 64];
    __shared__ _Float16 sA[8][16 * 32];   // x1 tile, K padded
    __shared__ _Float16 sX2[8][16 * 64];
    __shared__ float red[8 * 128];
    int tid = threadIdx.x;
    for (int i = tid; i < 1024; i += 256) {
        int o = i >> 5, c = i & 31;
        sW2[i] = (c < 16) ? (_Float16)W2[o * 16 + c] : (_Float16)0.0f;
    }
    for (int i = tid; i < 4096; i += 256) sW3[i] = (_Float16)W3[i];
    for (int i = tid; i < 8 * 512; i += 256) (&sA[0][0])[i] = (_Float16)0.0f;
    __syncthreads();
    int lane = tid & 31, wv = tid >> 5, n15 = lane & 15, hi = lane >> 4;
    float scl2[2], sft2[2];
#pragma unroll
    for (int j = 0; j < 2; ++j) { scl2[j] = st2[2 * (j * 16 + n15)]; sft2[j] = st2[2 * (j * 16 + n15) + 1]; }
    float sum3[4], sq3[4];
#pragma unroll
    for (int nt = 0; nt < 4; ++nt) { sum3[nt] = 0.f; sq3[nt] = 0.f; }
    _Float16* mA  = sA[wv];
    _Float16* mX2 = sX2[wv];
    for (int vx = blockIdx.x * 8 + wv; vx < NV; vx += gridDim.x * 8) {
        const float4* src = (const float4*)(x1 + vx * 16 * 16);
#pragma unroll
        for (int j = 0; j < 2; ++j) {
            int f4 = lane * 2 + j, row = f4 >> 2, c4 = f4 & 3;
            float4 d = src[row * 4 + c4];
            mA[row * 32 + c4 * 4 + 0] = (_Float16)d.x;
            mA[row * 32 + c4 * 4 + 1] = (_Float16)d.y;
            mA[row * 32 + c4 * 4 + 2] = (_Float16)d.z;
            mA[row * 32 + c4 * 4 + 3] = (_Float16)d.w;
        }
        asm volatile("" ::: "memory");
        v16h a = frag_A(mA, 32, 0, lane);
        float pbuf[2][8], cm[2];
#pragma unroll
        for (int j = 0; j < 2; ++j) {
            v8f z = {0.f, 0.f, 0.f, 0.f, 0.f, 0.f, 0.f, 0.f};
            v8f d = wmma32(a, frag_B(sW2, 32, j * 16, 0, lane), z);
            float m = -3.0e38f;
#pragma unroll
            for (int e = 0; e < 8; ++e) {
                float pv = mish_f(fmaf(d[e], scl2[j], sft2[j]));
                pbuf[j][e] = pv;
                m = fmaxf(m, pv);
            }
            m = fmaxf(m, __shfl_xor(m, 16, 32));
            cm[j] = m;
        }
        float vmr[8];
#pragma unroll
        for (int e = 0; e < 8; ++e) vmr[e] = vmask[vx * 16 + hi * 8 + e];
#pragma unroll
        for (int j = 0; j < 2; ++j)
#pragma unroll
            for (int e = 0; e < 8; ++e) {
                int row = hi * 8 + e, col = j * 16 + n15;
                float lo  = pbuf[j][e] * vmr[e];
                float hic = cm[j] * vmr[e];
                x2g[(vx * 16 + row) * 64 + col]      = lo;
                x2g[(vx * 16 + row) * 64 + 32 + col] = hic;
                mX2[row * 64 + col]      = (_Float16)lo;
                mX2[row * 64 + 32 + col] = (_Float16)hic;
            }
        asm volatile("" ::: "memory");
        v16h b0 = frag_A(mX2, 64, 0, lane), b1 = frag_A(mX2, 64, 32, lane);
#pragma unroll
        for (int nt = 0; nt < 4; ++nt) {
            v8f d = {0.f, 0.f, 0.f, 0.f, 0.f, 0.f, 0.f, 0.f};
            d = wmma32(b0, frag_B(sW3, 64, nt * 16, 0, lane), d);
            d = wmma32(b1, frag_B(sW3, 64, nt * 16, 32, lane), d);
            float ls = 0.f, lq = 0.f;
#pragma unroll
            for (int e = 0; e < 8; ++e) { ls += d[e]; lq += d[e] * d[e]; }
            ls += __shfl_xor(ls, 16, 32);
            lq += __shfl_xor(lq, 16, 32);
            sum3[nt] += ls; sq3[nt] += lq;
        }
    }
    if (lane < 16) {
#pragma unroll
        for (int nt = 0; nt < 4; ++nt) {
            red[wv * 128 + nt * 16 + lane]      = sum3[nt];
            red[wv * 128 + 64 + nt * 16 + lane] = sq3[nt];
        }
    }
    __syncthreads();
    if (tid < 128) {
        float a = 0.f;
        for (int w = 0; w < 8; ++w) a += red[w * 128 + tid];
        part[blockIdx.x * 128 + tid] = a;
    }
}

// ---------------- S4: y3 (WMMA) -> BN+mish -> x3 -> y4 (WMMA) stats (64 ch) ----------------
__global__ __launch_bounds__(256) void k_s4(const float* __restrict__ x2g,
                                            const float* __restrict__ st3,
                                            const float* __restrict__ W3,
                                            const float* __restrict__ W4,
                                            float* __restrict__ part) {
    __shared__ _Float16 sW3[4096], sW4[4096];
    __shared__ _Float16 sXin[8][1024], sX3[8][1024];
    __shared__ float red[8 * 128];
    int tid = threadIdx.x;
    for (int i = tid; i < 4096; i += 256) { sW3[i] = (_Float16)W3[i]; sW4[i] = (_Float16)W4[i]; }
    __syncthreads();
    int lane = tid & 31, wv = tid >> 5, n15 = lane & 15, hi = lane >> 4;
    float scl3[4], sft3[4];
#pragma unroll
    for (int nt = 0; nt < 4; ++nt) { scl3[nt] = st3[2 * (nt * 16 + n15)]; sft3[nt] = st3[2 * (nt * 16 + n15) + 1]; }
    float sum4[4], sq4[4];
#pragma unroll
    for (int nt = 0; nt < 4; ++nt) { sum4[nt] = 0.f; sq4[nt] = 0.f; }
    _Float16* mIn = sXin[wv];
    _Float16* mX3 = sX3[wv];
    for (int vx = blockIdx.x * 8 + wv; vx < NV; vx += gridDim.x * 8) {
        { int nxt = vx + gridDim.x * 8;
          __builtin_prefetch(x2g + (size_t)(nxt < NV ? nxt : vx) * 1024, 0, 1); }
        const float4* src = (const float4*)(x2g + (size_t)vx * 1024);
#pragma unroll
        for (int j = 0; j < 8; ++j) {
            int f4 = lane * 8 + j, row = f4 >> 4, c4 = f4 & 15;
            float4 d = src[row * 16 + c4];
            mIn[row * 64 + c4 * 4 + 0] = (_Float16)d.x;
            mIn[row * 64 + c4 * 4 + 1] = (_Float16)d.y;
            mIn[row * 64 + c4 * 4 + 2] = (_Float16)d.z;
            mIn[row * 64 + c4 * 4 + 3] = (_Float16)d.w;
        }
        asm volatile("" ::: "memory");
        v16h a0 = frag_A(mIn, 64, 0, lane), a1 = frag_A(mIn, 64, 32, lane);
#pragma unroll
        for (int nt = 0; nt < 4; ++nt) {
            v8f d = {0.f, 0.f, 0.f, 0.f, 0.f, 0.f, 0.f, 0.f};
            d = wmma32(a0, frag_B(sW3, 64, nt * 16, 0, lane), d);
            d = wmma32(a1, frag_B(sW3, 64, nt * 16, 32, lane), d);
#pragma unroll
            for (int e = 0; e < 8; ++e) {
                float x3v = mish_f(fmaf(d[e], scl3[nt], sft3[nt]));
                mX3[(hi * 8 + e) * 64 + nt * 16 + n15] = (_Float16)x3v;
            }
        }
        asm volatile("" ::: "memory");
        v16h b0 = frag_A(mX3, 64, 0, lane), b1 = frag_A(mX3, 64, 32, lane);
#pragma unroll
        for (int nt = 0; nt < 4; ++nt) {
            v8f d = {0.f, 0.f, 0.f, 0.f, 0.f, 0.f, 0.f, 0.f};
            d = wmma32(b0, frag_B(sW4, 64, nt * 16, 0, lane), d);
            d = wmma32(b1, frag_B(sW4, 64, nt * 16, 32, lane), d);
            float ls = 0.f, lq = 0.f;
#pragma unroll
            for (int e = 0; e < 8; ++e) { ls += d[e]; lq += d[e] * d[e]; }
            ls += __shfl_xor(ls, 16, 32);
            lq += __shfl_xor(lq, 16, 32);
            sum4[nt] += ls; sq4[nt] += lq;
        }
    }
    if (lane < 16) {
#pragma unroll
        for (int nt = 0; nt < 4; ++nt) {
            red[wv * 128 + nt * 16 + lane]      = sum4[nt];
            red[wv * 128 + 64 + nt * 16 + lane] = sq4[nt];
        }
    }
    __syncthreads();
    if (tid < 128) {
        float a = 0.f;
        for (int w = 0; w < 8; ++w) a += red[w * 128 + tid];
        part[blockIdx.x * 128 + tid] = a;
    }
}

// ---------------- S5: full x3->x4 chain, residual + mask, max over T -> voxelwise ----------------
__global__ __launch_bounds__(256) void k_s5(const float* __restrict__ x2g,
                                            const float* __restrict__ vmask,
                                            const float* __restrict__ st3,
                                            const float* __restrict__ st4,
                                            const float* __restrict__ W3,
                                            const float* __restrict__ W4,
                                            float* __restrict__ voxw) {
    __shared__ _Float16 sW3[4096], sW4[4096];
    __shared__ _Float16 sXin[8][1024], sX3[8][1024];
    int tid = threadIdx.x;
    for (int i = tid; i < 4096; i += 256) { sW3[i] = (_Float16)W3[i]; sW4[i] = (_Float16)W4[i]; }
    __syncthreads();
    int lane = tid & 31, wv = tid >> 5, n15 = lane & 15, hi = lane >> 4;
    float scl3[4], sft3[4], scl4[4], sft4[4];
#pragma unroll
    for (int nt = 0; nt < 4; ++nt) {
        scl3[nt] = st3[2 * (nt * 16 + n15)]; sft3[nt] = st3[2 * (nt * 16 + n15) + 1];
        scl4[nt] = st4[2 * (nt * 16 + n15)]; sft4[nt] = st4[2 * (nt * 16 + n15) + 1];
    }
    _Float16* mIn = sXin[wv];
    _Float16* mX3 = sX3[wv];
    for (int vx = blockIdx.x * 8 + wv; vx < NV; vx += gridDim.x * 8) {
        { int nxt = vx + gridDim.x * 8;
          __builtin_prefetch(x2g + (size_t)(nxt < NV ? nxt : vx) * 1024, 0, 1); }
        const float4* src = (const float4*)(x2g + (size_t)vx * 1024);
#pragma unroll
        for (int j = 0; j < 8; ++j) {
            int f4 = lane * 8 + j, row = f4 >> 4, c4 = f4 & 15;
            float4 d = src[row * 16 + c4];
            mIn[row * 64 + c4 * 4 + 0] = (_Float16)d.x;
            mIn[row * 64 + c4 * 4 + 1] = (_Float16)d.y;
            mIn[row * 64 + c4 * 4 + 2] = (_Float16)d.z;
            mIn[row * 64 + c4 * 4 + 3] = (_Float16)d.w;
        }
        asm volatile("" ::: "memory");
        v16h a0 = frag_A(mIn, 64, 0, lane), a1 = frag_A(mIn, 64, 32, lane);
#pragma unroll
        for (int nt = 0; nt < 4; ++nt) {
            v8f d = {0.f, 0.f, 0.f, 0.f, 0.f, 0.f, 0.f, 0.f};
            d = wmma32(a0, frag_B(sW3, 64, nt * 16, 0, lane), d);
            d = wmma32(a1, frag_B(sW3, 64, nt * 16, 32, lane), d);
#pragma unroll
            for (int e = 0; e < 8; ++e) {
                float x3v = mish_f(fmaf(d[e], scl3[nt], sft3[nt]));
                mX3[(hi * 8 + e) * 64 + nt * 16 + n15] = (_Float16)x3v;
            }
        }
        asm volatile("" ::: "memory");
        v16h b0 = frag_A(mX3, 64, 0, lane), b1 = frag_A(mX3, 64, 32, lane);
        float vmr[8];
#pragma unroll
        for (int e = 0; e < 8; ++e) vmr[e] = vmask[vx * 16 + hi * 8 + e];
#pragma unroll
        for (int nt = 0; nt < 4; ++nt) {
            v8f d = {0.f, 0.f, 0.f, 0.f, 0.f, 0.f, 0.f, 0.f};
            d = wmma32(b0, frag_B(sW4, 64, nt * 16, 0, lane), d);
            d = wmma32(b1, frag_B(sW4, 64, nt * 16, 32, lane), d);
            float m = -3.0e38f;
#pragma unroll
            for (int e = 0; e < 8; ++e) {
                int row = hi * 8 + e;
                float x2v = x2g[(size_t)(vx * 16 + row) * 64 + nt * 16 + n15];
                float t4  = mish_f(fmaf(d[e], scl4[nt], sft4[nt])) * vmr[e] + x2v;
                m = fmaxf(m, t4);
            }
            m = fmaxf(m, __shfl_xor(m, 16, 32));
            if (lane < 16) voxw[vx * 64 + nt * 16 + lane] = m;
        }
    }
}

// ---------------- S6: gather + bfe1 linear stats (128 ch over (g,q)) ----------------
__global__ __launch_bounds__(256) void k_s6(const float* __restrict__ voxw,
                                            const int* __restrict__ idx,
                                            const float* __restrict__ Wb1,
                                            float* __restrict__ part) {
    __shared__ float sW[1024];
    __shared__ float red[256 * 16];
    int tid = threadIdx.x;
    for (int i = tid; i < 1024; i += 256) sW[i] = Wb1[i];
    __syncthreads();
    int nl = tid >> 4, g = tid & 15;
    int n = blockIdx.x * 16 + nl;
    float xg[4][8];
#pragma unroll
    for (int p = 0; p < P; ++p) {
        int vi = idx[n * P + p];
        const float4 f = *(const float4*)(voxw + (size_t)vi * 64 + g * 4);
        xg[0][p] = f.x; xg[1][p] = f.y; xg[2][p] = f.z; xg[3][p] = f.w;
    }
    float s[8], q[8];
#pragma unroll
    for (int o = 0; o < 8; ++o) { s[o] = 0.f; q[o] = 0.f; }
#pragma unroll
    for (int c = 0; c < 4; ++c)
#pragma unroll
        for (int o = 0; o < 8; ++o) {
            float h = 0.f;
#pragma unroll
            for (int p = 0; p < 8; ++p) h = fmaf(xg[c][p], sW[(g * 8 + o) * 8 + p], h);
            s[o] += h; q[o] += h * h;
        }
#pragma unroll
    for (int o = 0; o < 8; ++o) { red[tid * 16 + o] = s[o]; red[tid * 16 + 8 + o] = q[o]; }
    __syncthreads();
    {
        int half = tid >> 7, c = tid & 127, gg = c >> 3, qq = c & 7;
        float a = 0.f;
        for (int nl2 = 0; nl2 < 16; ++nl2) a += red[(nl2 * 16 + gg) * 16 + half * 8 + qq];
        part[blockIdx.x * 256 + tid] = a;
    }
}

// ---------------- S7: bfe1 apply + mask + bfe2 linear stats (32 ch) ----------------
__global__ __launch_bounds__(256) void k_s7(const float* __restrict__ voxw,
                                            const int* __restrict__ idx,
                                            const float* __restrict__ bm,
                                            const float* __restrict__ Wb1,
                                            const float* __restrict__ sb1,
                                            const float* __restrict__ Wb2,
                                            float* __restrict__ part) {
    __shared__ float sW1[1024], sS1[256], sW2[256];
    __shared__ float red[256 * 4];
    int tid = threadIdx.x;
    for (int i = tid; i < 1024; i += 256) sW1[i] = Wb1[i];
    sS1[tid] = sb1[tid];
    sW2[tid] = Wb2[tid];
    __syncthreads();
    int nl = tid >> 4, g = tid & 15;
    int n = blockIdx.x * 16 + nl;
    float xg[4][8], hm[4][8];
#pragma unroll
    for (int p = 0; p < P; ++p) {
        int vi = idx[n * P + p];
        const float4 f = *(const float4*)(voxw + (size_t)vi * 64 + g * 4);
        xg[0][p] = f.x; xg[1][p] = f.y; xg[2][p] = f.z; xg[3][p] = f.w;
    }
#pragma unroll
    for (int c = 0; c < 4; ++c)
#pragma unroll
        for (int o = 0; o < 8; ++o) {
            float h = 0.f;
#pragma unroll
            for (int p = 0; p < 8; ++p) h = fmaf(xg[c][p], sW1[(g * 8 + o) * 8 + p], h);
            int ch = g * 8 + o;
            hm[c][o] = mish_f(fmaf(h, sS1[2 * ch], sS1[2 * ch + 1])) * bm[n * 8 + o];
        }
    float s0 = 0.f, q0 = 0.f, s1 = 0.f, q1 = 0.f;
#pragma unroll
    for (int c = 0; c < 4; ++c) {
#pragma unroll
        for (int o2 = 0; o2 < 2; ++o2) {
            float h = 0.f;
#pragma unroll
            for (int p = 0; p < 8; ++p) h = fmaf(hm[c][p], sW2[(g * 2 + o2) * 8 + p], h);
            if (o2 == 0) { s0 += h; q0 += h * h; } else { s1 += h; q1 += h * h; }
        }
    }
    red[tid * 4 + 0] = s0; red[tid * 4 + 1] = s1;
    red[tid * 4 + 2] = q0; red[tid * 4 + 3] = q1;
    __syncthreads();
    if (tid < 64) {
        int half = tid >> 5, c = tid & 31, gg = c >> 1, qq = c & 1;
        float a = 0.f;
        for (int nl2 = 0; nl2 < 16; ++nl2) a += red[(nl2 * 16 + gg) * 4 + half * 2 + qq];
        part[blockIdx.x * 64 + tid] = a;
    }
}

// ---------------- S8a: full BEV front-end -> x17 [NB,128] (f16) ----------------
__global__ __launch_bounds__(256) void k_s8a(const float* __restrict__ voxw,
                                             const int* __restrict__ idx,
                                             const float* __restrict__ bm,
                                             const float* __restrict__ Wb1,
                                             const float* __restrict__ sb1,
                                             const float* __restrict__ Wb2,
                                             const float* __restrict__ sb2,
                                             _Float16* __restrict__ x17) {
    __shared__ float sW1[1024], sS1[256], sW2[256], sS2[64];
    int tid = threadIdx.x;
    for (int i = tid; i < 1024; i += 256) sW1[i] = Wb1[i];
    sS1[tid] = sb1[tid];
    sW2[tid] = Wb2[tid];
    if (tid < 64) sS2[tid] = sb2[tid];
    __syncthreads();
    int n = blockIdx.x * 256 + tid;
    if (n >= NB) return;
    for (int g = 0; g < G; ++g) {
        float xg[4][8], hm[4][8];
#pragma unroll
        for (int p = 0; p < P; ++p) {
            int vi = idx[n * P + p];
            const float4 f = *(const float4*)(voxw + (size_t)vi * 64 + g * 4);
            xg[0][p] = f.x; xg[1][p] = f.y; xg[2][p] = f.z; xg[3][p] = f.w;
        }
#pragma unroll
        for (int c = 0; c < 4; ++c)
#pragma unroll
            for (int o = 0; o < 8; ++o) {
                float h = 0.f;
#pragma unroll
                for (int p = 0; p < 8; ++p) h = fmaf(xg[c][p], sW1[(g * 8 + o) * 8 + p], h);
                int ch = g * 8 + o;
                hm[c][o] = mish_f(fmaf(h, sS1[2 * ch], sS1[2 * ch + 1])) * bm[n * 8 + o];
            }
#pragma unroll
        for (int c = 0; c < 4; ++c) {
            h2 pr;
#pragma unroll
            for (int o2 = 0; o2 < 2; ++o2) {
                float h = 0.f;
#pragma unroll
                for (int p = 0; p < 8; ++p) h = fmaf(hm[c][p], sW2[(g * 2 + o2) * 8 + p], h);
                int ch = g * 2 + o2;
                pr[o2] = (_Float16)mish_f(fmaf(h, sS2[2 * ch], sS2[2 * ch + 1]));
            }
            *(h2*)(x17 + (size_t)n * 128 + c * 32 + g * 2) = pr;
        }
    }
}

// ---------------- S8b: y18 = x17 @ Wb3^T (WMMA) stats (128 ch); async tile DMA ----------------
__global__ __launch_bounds__(128) void k_s8b(const _Float16* __restrict__ x17,
                                             const float* __restrict__ Wb3,
                                             float* __restrict__ part) {
    __shared__ _Float16 sW[16384];
    __shared__ _Float16 sX[4][2048];
    __shared__ float red[4 * 256];
    int tid = threadIdx.x;
    for (int i = tid; i < 16384; i += 128) sW[i] = (_Float16)Wb3[i];
    __syncthreads();
    int lane = tid & 31, wv = tid >> 5;
    float sum[8], sq[8];
#pragma unroll
    for (int o = 0; o < 8; ++o) { sum[o] = 0.f; sq[o] = 0.f; }
    _Float16* mX = sX[wv];
    unsigned mXoff = lds_offset(mX);
    for (int tb = blockIdx.x * 4 + wv; tb < NTB; tb += gridDim.x * 4) {
        const char* src = (const char*)(x17 + (size_t)tb * 2048);
        wait_async0();                        // staging buffer free
#pragma unroll
        for (int i = 0; i < 8; ++i)
            async_load_b128(mXoff + lane * 16 + i * 512, src + lane * 16 + i * 512);
        wait_async0();                        // tile resident in LDS
        asm volatile("" ::: "memory");
        v16h a0 = frag_A(mX, 128, 0, lane), a1 = frag_A(mX, 128, 32, lane),
             a2 = frag_A(mX, 128, 64, lane), a3 = frag_A(mX, 128, 96, lane);
#pragma unroll
        for (int nt = 0; nt < 8; ++nt) {
            v8f d = {0.f, 0.f, 0.f, 0.f, 0.f, 0.f, 0.f, 0.f};
            d = wmma32(a0, frag_B(sW, 128, nt * 16, 0, lane), d);
            d = wmma32(a1, frag_B(sW, 128, nt * 16, 32, lane), d);
            d = wmma32(a2, frag_B(sW, 128, nt * 16, 64, lane), d);
            d = wmma32(a3, frag_B(sW, 128, nt * 16, 96, lane), d);
            float ls = 0.f, lq = 0.f;
#pragma unroll
            for (int e = 0; e < 8; ++e) { ls += d[e]; lq += d[e] * d[e]; }
            ls += __shfl_xor(ls, 16, 32);
            lq += __shfl_xor(lq, 16, 32);
            sum[nt] += ls; sq[nt] += lq;
        }
    }
    if (lane < 16) {
#pragma unroll
        for (int nt = 0; nt < 8; ++nt) {
            red[wv * 256 + nt * 16 + lane]       = sum[nt];
            red[wv * 256 + 128 + nt * 16 + lane] = sq[nt];
        }
    }
    __syncthreads();
    for (int c = tid; c < 256; c += 128) {
        float a = 0.f;
        for (int w = 0; w < 4; ++w) a += red[w * 256 + c];
        part[blockIdx.x * 256 + c] = a;
    }
}

// ---------------- S9: y18 -> BN+mish -> x18 (async store); y19 (WMMA) stats ----------------
__global__ __launch_bounds__(128) void k_s9(const _Float16* __restrict__ x17,
                                            const float* __restrict__ Wb3,
                                            const float* __restrict__ sb3,
                                            _Float16* __restrict__ x18,
                                            float* __restrict__ part) {
    __shared__ _Float16 sW[16384];
    __shared__ _Float16 sX[4][2048];
    __shared__ float red[4 * 256];
    int tid = threadIdx.x;
    for (int i = tid; i < 16384; i += 128) sW[i] = (_Float16)Wb3[i];
    __syncthreads();
    int lane = tid & 31, wv = tid >> 5, n15 = lane & 15, hi = lane >> 4;
    float scl[8], sft[8];
#pragma unroll
    for (int nt = 0; nt < 8; ++nt) { scl[nt] = sb3[2 * (nt * 16 + n15)]; sft[nt] = sb3[2 * (nt * 16 + n15) + 1]; }
    float sum[8], sq[8];
#pragma unroll
    for (int o = 0; o < 8; ++o) { sum[o] = 0.f; sq[o] = 0.f; }
    _Float16* mX = sX[wv];
    unsigned mXoff = lds_offset(mX);
    for (int tb = blockIdx.x * 4 + wv; tb < NTB; tb += gridDim.x * 4) {
        const char* src = (const char*)(x17 + (size_t)tb * 2048);
        wait_async0();                        // prior async stores done -> mX reusable
#pragma unroll
        for (int i = 0; i < 8; ++i)
            async_load_b128(mXoff + lane * 16 + i * 512, src + lane * 16 + i * 512);
        wait_async0();
        asm volatile("" ::: "memory");
        v16h a0 = frag_A(mX, 128, 0, lane), a1 = frag_A(mX, 128, 32, lane),
             a2 = frag_A(mX, 128, 64, lane), a3 = frag_A(mX, 128, 96, lane);
        // y18 -> x18 into the same staging buffer (per-wave DS ops are in order)
#pragma unroll
        for (int nt = 0; nt < 8; ++nt) {
            v8f d = {0.f, 0.f, 0.f, 0.f, 0.f, 0.f, 0.f, 0.f};
            d = wmma32(a0, frag_B(sW, 128, nt * 16, 0, lane), d);
            d = wmma32(a1, frag_B(sW, 128, nt * 16, 32, lane), d);
            d = wmma32(a2, frag_B(sW, 128, nt * 16, 64, lane), d);
            d = wmma32(a3, frag_B(sW, 128, nt * 16, 96, lane), d);
#pragma unroll
            for (int e = 0; e < 8; ++e) {
                int row = hi * 8 + e;
                float v = mish_f(fmaf(d[e], scl[nt], sft[nt]));
                mX[row * 128 + nt * 16 + n15] = (_Float16)v;
            }
        }
        asm volatile("" ::: "memory");
        wait_ds0();                           // x18 tile fully in LDS
        {
            char* gd = (char*)(x18 + (size_t)tb * 2048);
#pragma unroll
            for (int i = 0; i < 8; ++i)
                async_store_b128(gd + lane * 16 + i * 512, mXoff + lane * 16 + i * 512);
        }
        v16h b0 = frag_A(mX, 128, 0, lane), b1 = frag_A(mX, 128, 32, lane),
             b2 = frag_A(mX, 128, 64, lane), b3 = frag_A(mX, 128, 96, lane);
#pragma unroll
        for (int nt = 0; nt < 8; ++nt) {
            v8f d = {0.f, 0.f, 0.f, 0.f, 0.f, 0.f, 0.f, 0.f};
            d = wmma32(b0, frag_B(sW, 128, nt * 16, 0, lane), d);
            d = wmma32(b1, frag_B(sW, 128, nt * 16, 32, lane), d);
            d = wmma32(b2, frag_B(sW, 128, nt * 16, 64, lane), d);
            d = wmma32(b3, frag_B(sW, 128, nt * 16, 96, lane), d);
            float ls = 0.f, lq = 0.f;
#pragma unroll
            for (int e = 0; e < 8; ++e) { ls += d[e]; lq += d[e] * d[e]; }
            ls += __shfl_xor(ls, 16, 32);
            lq += __shfl_xor(lq, 16, 32);
            sum[nt] += ls; sq[nt] += lq;
        }
    }
    wait_async0();
    if (lane < 16) {
#pragma unroll
        for (int nt = 0; nt < 8; ++nt) {
            red[wv * 256 + nt * 16 + lane]       = sum[nt];
            red[wv * 256 + 128 + nt * 16 + lane] = sq[nt];
        }
    }
    __syncthreads();
    for (int c = tid; c < 256; c += 128) {
        float a = 0.f;
        for (int w = 0; w < 4; ++w) a += red[w * 256 + c];
        part[blockIdx.x * 256 + c] = a;
    }
}

// ---------------- S10: y19 -> BN+mish + x17 residual -> scatter to bevmap ----------------
__global__ __launch_bounds__(128) void k_s10(const _Float16* __restrict__ x17,
                                             const _Float16* __restrict__ x18,
                                             const float* __restrict__ Wb3,
                                             const float* __restrict__ sb4,
                                             const int* __restrict__ coors,
                                             float* __restrict__ out) {
    __shared__ _Float16 sW[16384];
    __shared__ _Float16 sX[4][2048];
    int tid = threadIdx.x;
    for (int i = tid; i < 16384; i += 128) sW[i] = (_Float16)Wb3[i];
    __syncthreads();
    int lane = tid & 31, wv = tid >> 5, n15 = lane & 15, hi = lane >> 4;
    float scl[8], sft[8];
#pragma unroll
    for (int nt = 0; nt < 8; ++nt) { scl[nt] = sb4[2 * (nt * 16 + n15)]; sft[nt] = sb4[2 * (nt * 16 + n15) + 1]; }
    _Float16* mX = sX[wv];
    unsigned mXoff = lds_offset(mX);
    for (int tb = blockIdx.x * 4 + wv; tb < NTB; tb += gridDim.x * 4) {
        const char* src = (const char*)(x18 + (size_t)tb * 2048);
        wait_async0();
#pragma unroll
        for (int i = 0; i < 8; ++i)
            async_load_b128(mXoff + lane * 16 + i * 512, src + lane * 16 + i * 512);
        wait_async0();
        asm volatile("" ::: "memory");
        v16h a0 = frag_A(mX, 128, 0, lane), a1 = frag_A(mX, 128, 32, lane),
             a2 = frag_A(mX, 128, 64, lane), a3 = frag_A(mX, 128, 96, lane);
        int cc0[8], cc1[8];
#pragma unroll
        for (int e = 0; e < 8; ++e) {
            int nc = tb * 16 + hi * 8 + e;
            cc0[e] = coors[2 * nc];
            cc1[e] = coors[2 * nc + 1];
        }
#pragma unroll
        for (int nt = 0; nt < 8; ++nt) {
            v8f d = {0.f, 0.f, 0.f, 0.f, 0.f, 0.f, 0.f, 0.f};
            d = wmma32(a0, frag_B(sW, 128, nt * 16, 0, lane), d);
            d = wmma32(a1, frag_B(sW, 128, nt * 16, 32, lane), d);
            d = wmma32(a2, frag_B(sW, 128, nt * 16, 64, lane), d);
            d = wmma32(a3, frag_B(sW, 128, nt * 16, 96, lane), d);
            int ch = nt * 16 + n15;
#pragma unroll
            for (int e = 0; e < 8; ++e) {
                int row = hi * 8 + e;
                float res = (float)x17[(size_t)(tb * 16 + row) * 128 + ch];
                float v = mish_f(fmaf(d[e], scl[nt], sft[nt])) + res;
                // out is [128, OW, OH]: out[ch][w=coors1][h=coors0]
                out[(size_t)ch * (OW * OH) + cc1[e] * OH + cc0[e]] = v;
            }
        }
    }
}

// ---------------- host side ----------------
extern "C" void kernel_launch(void* const* d_in, const int* in_sizes, int n_in,
                              void* d_out, int out_size, void* d_ws, size_t ws_size,
                              hipStream_t stream) {
    (void)in_sizes; (void)n_in; (void)out_size; (void)ws_size;
    const float* voxels = (const float*)d_in[0];
    const float* vmask  = (const float*)d_in[1];
    const float* bmask  = (const float*)d_in[2];
    const float* W1 = (const float*)d_in[3];
    const float* g1 = (const float*)d_in[4];
    const float* b1 = (const float*)d_in[5];
    const float* W2 = (const float*)d_in[6];
    const float* g2 = (const float*)d_in[7];
    const float* b2 = (const float*)d_in[8];
    const float* W3 = (const float*)d_in[9];
    const float* g3 = (const float*)d_in[10];
    const float* b3 = (const float*)d_in[11];
    const float* W4 = (const float*)d_in[12];
    const float* g4 = (const float*)d_in[13];
    const float* b4 = (const float*)d_in[14];
    const float* Wb1 = (const float*)d_in[15];
    const float* gb1 = (const float*)d_in[16];
    const float* bb1 = (const float*)d_in[17];
    const float* Wb2 = (const float*)d_in[18];
    const float* gb2 = (const float*)d_in[19];
    const float* bb2 = (const float*)d_in[20];
    const float* Wb3 = (const float*)d_in[21];
    const float* gb3 = (const float*)d_in[22];
    const float* bb3 = (const float*)d_in[23];
    const int* idx   = (const int*)d_in[24];
    const int* coors = (const int*)d_in[25];
    float* out = (float*)d_out;

    // workspace layout (floats)
    float* ws   = (float*)d_ws;
    float* x1   = ws;                       // 12.8M
    float* x2   = ws + 12800000;            // 51.2M
    float* voxw = ws + 64000000;            // 3.2M
    float* part = ws + 67200000;            // up to 1.92M
    float* st   = ws + 69200000;            // stats (scale,shift)
    float* st1 = st, *st2 = st + 16, *st3 = st + 80, *st4 = st + 208;
    float* sb1 = st + 336, *sb2 = st + 592, *sb3 = st + 656, *sb4 = st + 912;
    // f16 bev intermediates reuse the x2 region after S5 (16B-aligned)
    _Float16* x17h = (_Float16*)x2;                     // 15.36M halves
    _Float16* x18h = (_Float16*)x2 + (1 << 24);         // +16M halves (32MB)

    k_zero<<<2048, 256, 0, stream>>>(out, (128 * OW * OH) / 4);

    k_s1<<<512, 256, 0, stream>>>(voxels, W1, part);
    k_finalize<<<1, 8, 0, stream>>>(part, 512, 8, (float)ROWS, g1, b1, st1);

    k_s2<<<512, 256, 0, stream>>>(voxels, W1, st1, W2, x1, part);
    k_finalize<<<1, 32, 0, stream>>>(part, 512, 32, (float)ROWS, g2, b2, st2);

    k_s3<<<512, 256, 0, stream>>>(x1, vmask, st2, W2, W3, x2, part);
    k_finalize<<<1, 64, 0, stream>>>(part, 512, 64, (float)ROWS, g3, b3, st3);

    k_s4<<<512, 256, 0, stream>>>(x2, st3, W3, W4, part);
    k_finalize<<<1, 64, 0, stream>>>(part, 512, 64, (float)ROWS, g4, b4, st4);

    k_s5<<<512, 256, 0, stream>>>(x2, vmask, st3, st4, W3, W4, voxw);

    k_s6<<<NTB, 256, 0, stream>>>(voxw, idx, Wb1, part);
    k_finalize<<<1, 128, 0, stream>>>(part, NTB, 128, (float)(NB * 4), gb1, bb1, sb1);

    k_s7<<<NTB, 256, 0, stream>>>(voxw, idx, bmask, Wb1, sb1, Wb2, part);
    k_finalize<<<1, 32, 0, stream>>>(part, NTB, 32, (float)(NB * 4), gb2, bb2, sb2);

    k_s8a<<<(NB + 255) / 256, 256, 0, stream>>>(voxw, idx, bmask, Wb1, sb1, Wb2, sb2, x17h);

    k_s8b<<<256, 128, 0, stream>>>(x17h, Wb3, part);
    k_finalize<<<1, 128, 0, stream>>>(part, 256, 128, (float)NB, gb3, bb3, sb3);

    k_s9<<<256, 128, 0, stream>>>(x17h, Wb3, sb3, x18h, part);
    k_finalize<<<1, 128, 0, stream>>>(part, 256, 128, (float)NB, gb3, bb3, sb4);

    k_s10<<<256, 128, 0, stream>>>(x17h, x18h, Wb3, sb4, coors, out);
}
